// CascadeImageGuideUpsample_59167469469726
// MI455X (gfx1250) — matrix operand order repaired and checked by reading, not compiled, not run on hardware
//
#include <hip/hip_runtime.h>
#include <hip/hip_bf16.h>

typedef __attribute__((ext_vector_type(16))) _Float16 v16h;
typedef __attribute__((ext_vector_type(8)))  _Float16 v8h;
typedef __attribute__((ext_vector_type(8)))  float    v8f;

#define CONV_TILE_N 128      // block N tile (4 waves x 32 columns)
#define LA_STRIDE   328      // 320 + 8 halves pad -> 656 B row stride (16B aligned)
#define LB_STRIDE   40       // 32 + 8 halves pad  -> 80 B row stride (16B aligned)

// ---------------------------------------------------------------------------
// 2x bilinear downsample (half-pixel, align_corners=False) == 2x2 average
__global__ void down2_kernel(const float* __restrict__ src, float* __restrict__ dst,
                             int C, int Hs, int Ws) {
    int Hd = Hs >> 1, Wd = Ws >> 1;
    int total = C * Hd * Wd;
    int idx = blockIdx.x * blockDim.x + threadIdx.x;
    if (idx >= total) return;
    int x = idx % Wd; int t = idx / Wd;
    int y = t % Hd;   int c = t / Hd;
    const float* p = src + ((size_t)c * Hs + 2 * y) * Ws + 2 * x;
    dst[idx] = 0.25f * (p[0] + p[1] + p[Ws] + p[Ws + 1]);
}

// ---------------------------------------------------------------------------
// out = 2*base - bilinear_up2(low).  low: [C,hl,wl], base/out: [C,2hl,2wl]
__global__ void up2_highboost_kernel(const float* __restrict__ low,
                                     const float* __restrict__ base,
                                     float* __restrict__ out,
                                     int C, int hl, int wl) {
    int H = 2 * hl, W = 2 * wl;
    int total = C * H * W;
    int idx = blockIdx.x * blockDim.x + threadIdx.x;
    if (idx >= total) return;
    int x = idx % W; int t = idx / W;
    int y = t % H;   int c = t / H;
    int ky = y >> 1; int oy = (y & 1) ? ky + 1 : ky - 1;
    oy = oy < 0 ? 0 : (oy > hl - 1 ? hl - 1 : oy);
    int kx = x >> 1; int ox = (x & 1) ? kx + 1 : kx - 1;
    ox = ox < 0 ? 0 : (ox > wl - 1 ? wl - 1 : ox);
    const float* L = low + (size_t)c * hl * wl;
    float v = 0.5625f * L[(size_t)ky * wl + kx]
            + 0.1875f * L[(size_t)ky * wl + ox]
            + 0.1875f * L[(size_t)oy * wl + kx]
            + 0.0625f * L[(size_t)oy * wl + ox];
    out[idx] = 2.0f * base[idx] - v;
}

// ---------------------------------------------------------------------------
// Pack conv weights [25][35][3][3] -> fp16 Wmat[32][320] (zero padded).
__global__ void build_wmat_kernel(const float* __restrict__ w, _Float16* __restrict__ Wmat) {
    int idx = blockIdx.x * blockDim.x + threadIdx.x;
    if (idx >= 32 * 320) return;
    int m = idx / 320, k = idx % 320;
    float v = (m < 25 && k < 315) ? w[m * 315 + k] : 0.0f;
    Wmat[idx] = (_Float16)v;
}

// ---------------------------------------------------------------------------
// Build fp16 concat tensor with 1-px zero halo:
// Xp[35][H2+2][W2+2]; ch 0..31 = nearest2x(feat) (implicit y>>1), ch 32..34 = img_hf.
__global__ void build_xp_kernel(const float* __restrict__ feat, int Hf, int Wf,
                                const float* __restrict__ imghf,
                                _Float16* __restrict__ Xp, int H2, int W2) {
    int Hp = H2 + 2, Wp = W2 + 2;
    int total = 35 * Hp * Wp;
    int idx = blockIdx.x * blockDim.x + threadIdx.x;
    if (idx >= total) return;
    int xp = idx % Wp; int t = idx / Wp;
    int yp = t % Hp;   int c = t / Hp;
    int y = yp - 1, x = xp - 1;
    float v = 0.0f;
    if (y >= 0 && y < H2 && x >= 0 && x < W2) {
        if (c < 32) v = feat[((size_t)c * Hf + (y >> 1)) * Wf + (x >> 1)];
        else        v = imghf[((size_t)(c - 32) * H2 + y) * W2 + x];
    }
    Xp[idx] = (_Float16)v;
}

// ---------------------------------------------------------------------------
// Implicit-im2col GEMM via WMMA:
//   logits[25 x S] = Wmat[32 x 320] * B[320 x S] + bias,  S = H2*W2
//   B[k][n] = Xp[c][y+di][x+dj], k = c*9 + di*3 + dj (haloed, no bounds check)
// Block = 128 threads = 4 waves; each wave owns 32 N columns (2 x 16-wide frags);
// block tile = 32(M) x 128(N); K loop = 10 chunks of 32, unrolled by 2.
__global__ __launch_bounds__(128) void conv_gemm_wmma_kernel(
        const _Float16* __restrict__ Xp, const _Float16* __restrict__ Wmat,
        const float* __restrict__ bias, float* __restrict__ out,
        int H2, int W2) {
    const int S  = H2 * W2;
    const int Wp = W2 + 2;
    const size_t chs = (size_t)(H2 + 2) * Wp;          // Xp channel stride

    __shared__ _Float16 lA[32 * LA_STRIDE];            // full 32x320 A (row-major, padded)
    __shared__ _Float16 lB[CONV_TILE_N * LB_STRIDE];   // B tile TRANSPOSED: [col][k]

    const int tid   = threadIdx.x;
    const int wave  = tid >> 5;
    const int lane  = tid & 31;
    const int lhalf = lane >> 4;                       // 0: lanes 0-15, 1: lanes 16-31
    const int l15   = lane & 15;
    const int n0    = blockIdx.x * CONV_TILE_N;

    // Load whole A into LDS once.
    for (int i = tid; i < 32 * 320; i += 128) {
        lA[(i / 320) * LA_STRIDE + (i % 320)] = Wmat[i];
    }

    v8f acc00 = {}, acc01 = {}, acc10 = {}, acc11 = {};

    const int colA = n0 + (wave << 5) + l15;           // this lane's N for frag 0
    const int colB = colA + 16;                        // and for frag 1

#pragma unroll 2
    for (int ck = 0; ck < 10; ++ck) {
        __syncthreads();
        // Fill B tile transposed: 32 K rows x 128 N cols; kk-major so consecutive
        // lanes gather consecutive x (coalesced global_load_u16).
        for (int i = tid; i < 32 * CONV_TILE_N; i += 128) {
            int col = i & (CONV_TILE_N - 1);
            int kk  = i >> 7;                          // log2(CONV_TILE_N)
            int k   = ck * 32 + kk;
            int n   = n0 + col;
            _Float16 v = (_Float16)0.0f;
            if (k < 315 && n < S) {
                int c  = k / 9;
                int t9 = k - c * 9;
                int di = t9 / 3, dj = t9 - di * 3;
                int y  = n / W2, x = n - y * W2;
                const _Float16* src = &Xp[(size_t)c * chs + (size_t)(y + di) * Wp + (x + dj)];
                v = *src;
                // prefetch next K-chunk's source line (gfx1250 global_prefetch_b8)
                int kn = k + 32;
                if (kn < 315) {
                    int cn  = kn / 9;
                    int t9n = kn - cn * 9;
                    int din = t9n / 3, djn = t9n - din * 3;
                    __builtin_prefetch(&Xp[(size_t)cn * chs + (size_t)(y + din) * Wp + (x + djn)], 0, 1);
                }
            }
            lB[col * LB_STRIDE + kk] = v;
        }
        __syncthreads();

        // A frags: row l15 (tile 0) / 16+l15 (tile 1); K = {0..7,16..23} + 8*lhalf.
        const _Float16* a0p = &lA[l15 * LA_STRIDE + ck * 32 + (lhalf << 3)];
        const _Float16* a1p = a0p + 16 * LA_STRIDE;
        v8h a0lo = *(const v8h*)(a0p);
        v8h a0hi = *(const v8h*)(a0p + 16);
        v8h a1lo = *(const v8h*)(a1p);
        v8h a1hi = *(const v8h*)(a1p + 16);
        v16h a0 = __builtin_shufflevector(a0lo, a0hi, 0,1,2,3,4,5,6,7,8,9,10,11,12,13,14,15);
        v16h a1 = __builtin_shufflevector(a1lo, a1hi, 0,1,2,3,4,5,6,7,8,9,10,11,12,13,14,15);

        // B frags: K = {0..7,16..23} + 8*lhalf in transposed tile (matches A).
        const _Float16* b0p = &lB[(colA - n0) * LB_STRIDE + (lhalf << 3)];
        const _Float16* b1p = &lB[(colB - n0) * LB_STRIDE + (lhalf << 3)];
        v8h b0lo = *(const v8h*)(b0p);
        v8h b0hi = *(const v8h*)(b0p + 16);
        v8h b1lo = *(const v8h*)(b1p);
        v8h b1hi = *(const v8h*)(b1p + 16);
        v16h b0 = __builtin_shufflevector(b0lo, b0hi, 0,1,2,3,4,5,6,7,8,9,10,11,12,13,14,15);
        v16h b1 = __builtin_shufflevector(b1lo, b1hi, 0,1,2,3,4,5,6,7,8,9,10,11,12,13,14,15);

        acc00 = __builtin_amdgcn_wmma_f32_16x16x32_f16(false, a0, false, b0, (short)0, acc00, false, false);
        acc01 = __builtin_amdgcn_wmma_f32_16x16x32_f16(false, a0, false, b1, (short)0, acc01, false, false);
        acc10 = __builtin_amdgcn_wmma_f32_16x16x32_f16(false, a1, false, b0, (short)0, acc10, false, false);
        acc11 = __builtin_amdgcn_wmma_f32_16x16x32_f16(false, a1, false, b1, (short)0, acc11, false, false);
    }

    // Store: C/D layout -> VGPR i holds M = i + 8*lhalf (per 16-row tile), N = l15.
#pragma unroll
    for (int i = 0; i < 8; ++i) {
        int m0 = i + (lhalf << 3);                     // rows 0..15: always valid logits
        int m1 = 16 + i + (lhalf << 3);                // rows 16..31: valid if < 25
        if (colA < S) {
            out[(size_t)m0 * S + colA] = acc00[i] + bias[m0];
            if (m1 < 25) out[(size_t)m1 * S + colA] = acc10[i] + bias[m1];
        }
        if (colB < S) {
            out[(size_t)m0 * S + colB] = acc01[i] + bias[m0];
            if (m1 < 25) out[(size_t)m1 * S + colB] = acc11[i] + bias[m1];
        }
    }
}

// ---------------------------------------------------------------------------
// Fused channel-softmax (25 taps) + CARAFE with implicit nearest2x of src.
__global__ void softmax_carafe_kernel(const float* __restrict__ mask,
                                      const float* __restrict__ src,
                                      int C, int Hs, int Ws,
                                      float* __restrict__ out, int H2, int W2) {
    int S = H2 * W2;
    int n = blockIdx.x * blockDim.x + threadIdx.x;
    if (n >= S) return;
    int y = n / W2, x = n - y * W2;

    float m[25];
    float mx = -1e30f;
#pragma unroll
    for (int i = 0; i < 25; ++i) { m[i] = mask[(size_t)i * S + n]; mx = fmaxf(mx, m[i]); }
    float s = 0.0f;
#pragma unroll
    for (int i = 0; i < 25; ++i) { m[i] = __expf(m[i] - mx); s += m[i]; }
    float inv = 1.0f / s;
#pragma unroll
    for (int i = 0; i < 25; ++i) m[i] *= inv;

    for (int c = 0; c < C; ++c) {
        const float* sp = src + (size_t)c * Hs * Ws;
        float acc = 0.0f;
        int i = 0;
#pragma unroll
        for (int di = 0; di < 5; ++di) {
            int yy = y + di - 2;
            bool vy = (yy >= 0 && yy < H2);
            int ys = yy >> 1;
#pragma unroll
            for (int dj = 0; dj < 5; ++dj, ++i) {
                int xx = x + dj - 2;
                if (vy && xx >= 0 && xx < W2)
                    acc += sp[(size_t)ys * Ws + (xx >> 1)] * m[i];
            }
        }
        out[(size_t)c * S + n] = acc;
    }
}

// ---------------------------------------------------------------------------
static inline int ceil_div(int a, int b) { return (a + b - 1) / b; }

extern "C" void kernel_launch(void* const* d_in, const int* in_sizes, int n_in,
                              void* d_out, int out_size, void* d_ws, size_t ws_size,
                              hipStream_t stream) {
    (void)in_sizes; (void)n_in; (void)out_size; (void)ws_size;

    const float* feat0  = (const float*)d_in[0];   // [32,96,96]
    const float* seg0   = (const float*)d_in[1];   // [19,96,96]
    const float* img    = (const float*)d_in[2];   // [3,384,384]
    const float* w_enc0 = (const float*)d_in[3];   // [25,35,3,3]
    const float* b_enc0 = (const float*)d_in[4];   // [25]
    const float* w_enc1 = (const float*)d_in[5];
    const float* b_enc1 = (const float*)d_in[6];
    float* segout = (float*)d_out;                 // [19,384,384]

    // Workspace layout (256B aligned)
    char* ws = (char*)d_ws;
    size_t off = 0;
    auto take = [&](size_t bytes) { size_t r = off; off = (off + bytes + 255) & ~(size_t)255; return r; };
    _Float16* Xp    = (_Float16*)(ws + take((size_t)35 * 386 * 386 * 2)); // concat, haloed, fp16
    float*    logit = (float*)   (ws + take((size_t)25 * 384 * 384 * 4)); // conv output
    float*    feat1 = (float*)   (ws + take((size_t)32 * 192 * 192 * 4));
    float*    seg1  = (float*)   (ws + take((size_t)19 * 192 * 192 * 4));
    float*    tmpA  = (float*)   (ws + take((size_t)3  * 192 * 192 * 4)); // down2(img)
    float*    tmpB  = (float*)   (ws + take((size_t)3  *  96 *  96 * 4)); // down2(down2(img))
    float*    imghf = (float*)   (ws + take((size_t)3  * 384 * 384 * 4));
    _Float16* Wmat  = (_Float16*)(ws + take((size_t)32 * 320 * 2));

    const int B = 256;

    // ===================== Level 0: 96 -> 192 =====================
    {
        const int Hf = 96, H2 = 192, S = H2 * H2;
        down2_kernel<<<ceil_div(3 * 192 * 192, B), B, 0, stream>>>(img, tmpA, 3, 384, 384);
        down2_kernel<<<ceil_div(3 * 96 * 96, B), B, 0, stream>>>(tmpA, tmpB, 3, 192, 192);
        up2_highboost_kernel<<<ceil_div(3 * 192 * 192, B), B, 0, stream>>>(tmpB, tmpA, imghf, 3, 96, 96);
        build_wmat_kernel<<<ceil_div(32 * 320, B), B, 0, stream>>>(w_enc0, Wmat);
        build_xp_kernel<<<ceil_div(35 * (H2 + 2) * (H2 + 2), B), B, 0, stream>>>(
            feat0, Hf, Hf, imghf, Xp, H2, H2);
        conv_gemm_wmma_kernel<<<ceil_div(S, CONV_TILE_N), 128, 0, stream>>>(
            Xp, Wmat, b_enc0, logit, H2, H2);
        softmax_carafe_kernel<<<ceil_div(S, B), B, 0, stream>>>(logit, seg0, 19, Hf, Hf, seg1, H2, H2);
        softmax_carafe_kernel<<<ceil_div(S, B), B, 0, stream>>>(logit, feat0, 32, Hf, Hf, feat1, H2, H2);
    }

    // ===================== Level 1: 192 -> 384 =====================
    {
        const int Hf = 192, H2 = 384, S = H2 * H2;
        // resized = img (identity resize); tmpA already holds down2(img)
        up2_highboost_kernel<<<ceil_div(3 * 384 * 384, B), B, 0, stream>>>(tmpA, img, imghf, 3, 192, 192);
        build_wmat_kernel<<<ceil_div(32 * 320, B), B, 0, stream>>>(w_enc1, Wmat);
        build_xp_kernel<<<ceil_div(35 * (H2 + 2) * (H2 + 2), B), B, 0, stream>>>(
            feat1, Hf, Hf, imghf, Xp, H2, H2);
        conv_gemm_wmma_kernel<<<ceil_div(S, CONV_TILE_N), 128, 0, stream>>>(
            Xp, Wmat, b_enc1, logit, H2, H2);
        softmax_carafe_kernel<<<ceil_div(S, B), B, 0, stream>>>(logit, seg1, 19, Hf, Hf, segout, H2, H2);
    }
}
